// PrimeKGDrugRepurposingGNN_39058432590091
// MI455X (gfx1250) — compile-verified, matrix operand order checked by reading, not run on hardware
//
#include <hip/hip_runtime.h>
#include <hip/hip_bf16.h>

typedef __attribute__((ext_vector_type(16))) __bf16 v16bf;
typedef __attribute__((ext_vector_type(8)))  float  v8f;

#define GNN_N 50000
#define GNN_E 800000
#define GNN_H 128
#define GNN_D 64
#define GNN_P 65536
// rows padded up to the 128-row block grid: 391 * 128
#define GNN_NPAD 50048

#define WMMA_BF16(a, b, c) \
    __builtin_amdgcn_wmma_f32_16x16x32_bf16(false, (a), false, (b), (short)0, (c), false, false)

// K index inside a 16-bit 16x32 A/B WMMA fragment for vector element i, lane l
__device__ __forceinline__ int kidx(int i, int lane) {
    return ((i >> 3) << 4) + (((lane >> 4) & 1) << 3) + (i & 7);
}

// ---------------------------------------------------------------- encode ----
__global__ void k_encode(const int* __restrict__ types,
                         const float* __restrict__ node_emb,
                         const float* __restrict__ type_emb,
                         float* __restrict__ X) {
    int idx = blockIdx.x * blockDim.x + threadIdx.x;
    if (idx >= GNN_N * GNN_H) return;
    int n = idx >> 7;
    int hcol = idx & 127;
    X[idx] = node_emb[idx] + type_emb[types[n] * GNN_H + hcol];
}

// ------------------------------------------------------------------ zero ----
__global__ void k_zero(float* __restrict__ p, int count) {
    int idx = blockIdx.x * blockDim.x + threadIdx.x;
    if (idx < count) p[idx] = 0.0f;
}

// ------------------------------------------- weight repack -> B fragments ----
__global__ void k_pack(const float* __restrict__ W, __bf16* __restrict__ P,
                       int K, int Ncols) {
    int ksteps = K >> 5;
    int total = (Ncols >> 4) * ksteps * 32 * 16;
    int idx = blockIdx.x * blockDim.x + threadIdx.x;
    if (idx >= total) return;
    int i    = idx & 15;
    int lane = (idx >> 4) & 31;
    int step = idx >> 9;            // ct*ksteps + ks
    int ct = step / ksteps;
    int ks = step - ct * ksteps;
    int k = ks * 32 + kidx(i, lane);
    int n = ct * 16 + (lane & 15);
    P[idx] = (__bf16)W[k * Ncols + n];
}

// --------------------------------------------------------- spmm (atomic) ----
__global__ void k_spmm(const int* __restrict__ rows,
                       const int* __restrict__ cols,
                       const float* __restrict__ vals,
                       const float* __restrict__ X,
                       float* __restrict__ Out) {
    int e = blockIdx.x * 8 + (threadIdx.x >> 5);
    if (e >= GNN_E) return;
    int lane = threadIdx.x & 31;
    int r = rows[e];
    int c = cols[e];
    float v = vals[e];
    const float* xr = X + (long)c * GNN_H;
    float* orow = Out + (long)r * GNN_H;
#pragma unroll
    for (int f = 0; f < GNN_H; f += 32)
        atomicAdd(&orow[f + lane], v * xr[f + lane]);
}

// ----------------------------------------- gemm1: [N,128]x[128,128]+relu ----
// wave owns 16 rows, A fragments resident, loops 8 col-tiles (32 WMMAs/wave).
// Buffers are padded to NPAD rows: no clamps, no store guards, EXEC all-1s.
__global__ void k_gemm1(const float* __restrict__ A,     // [NPAD,128]
                        const __bf16* __restrict__ Wp,
                        const float* __restrict__ bias,
                        float* __restrict__ Out) {       // [NPAD,128]
    int lane = threadIdx.x & 31;
    int wave = threadIdx.x >> 5;
    int rowBase = blockIdx.x * 128 + wave * 16;
    int m = lane & 15;
    const float* arow = A + (long)(rowBase + m) * GNN_H;

    v16bf afrag[4];
#pragma unroll
    for (int ks = 0; ks < 4; ++ks)
#pragma unroll
        for (int i = 0; i < 16; ++i)
            afrag[ks][i] = (__bf16)arow[ks * 32 + kidx(i, lane)];

    const v16bf* bfr = (const v16bf*)Wp;
#pragma unroll
    for (int ct = 0; ct < 8; ++ct) {
        v8f acc = {};
#pragma unroll
        for (int ks = 0; ks < 4; ++ks)
            acc = WMMA_BF16(afrag[ks], bfr[(ct * 4 + ks) * 32 + lane], acc);
        int n = ct * 16 + m;
        float bv = bias[n];
#pragma unroll
        for (int r = 0; r < 8; ++r) {
            int orow = rowBase + r + ((lane >> 4) << 3);
            float v = acc[r] + bv;
            Out[(long)orow * GNN_H + n] = v > 0.0f ? v : 0.0f;
        }
    }
}

// ------------------------------------------------ gemm2: [N,128]x[128,64] ----
__global__ void k_gemm2(const float* __restrict__ A,     // [NPAD,128]
                        const __bf16* __restrict__ Wp,
                        const float* __restrict__ bias,
                        float* __restrict__ Out) {       // [NPAD,64]
    int lane = threadIdx.x & 31;
    int wave = threadIdx.x >> 5;
    int rowBase = blockIdx.x * 128 + wave * 16;
    int m = lane & 15;
    const float* arow = A + (long)(rowBase + m) * GNN_H;

    v16bf afrag[4];
#pragma unroll
    for (int ks = 0; ks < 4; ++ks)
#pragma unroll
        for (int i = 0; i < 16; ++i)
            afrag[ks][i] = (__bf16)arow[ks * 32 + kidx(i, lane)];

    const v16bf* bfr = (const v16bf*)Wp;
#pragma unroll
    for (int ct = 0; ct < 4; ++ct) {
        v8f acc = {};
#pragma unroll
        for (int ks = 0; ks < 4; ++ks)
            acc = WMMA_BF16(afrag[ks], bfr[(ct * 4 + ks) * 32 + lane], acc);
        int n = ct * 16 + m;
        float bv = bias[n];
#pragma unroll
        for (int r = 0; r < 8; ++r) {
            int orow = rowBase + r + ((lane >> 4) << 3);
            Out[(long)orow * GNN_D + n] = acc[r] + bv;
        }
    }
}

// ------------------------------------------------------------ pair score ----
__global__ void k_pairs(const float* __restrict__ Z,     // [NPAD,64]
                        const int* __restrict__ pairs0,
                        const int* __restrict__ pairs1,
                        const __bf16* __restrict__ Wp,
                        const float* __restrict__ bp1,
                        const float* __restrict__ Wp2,
                        const float* __restrict__ bp2,
                        float* __restrict__ Out) {
    int lane = threadIdx.x & 31;
    int wave = threadIdx.x >> 5;
    int pairBase = blockIdx.x * 128 + wave * 16;   // P % 128 == 0: no guards
    int m = lane & 15;

    int p = pairBase + m;
    long srow = (long)pairs0[p] * GNN_D;
    long drow = (long)pairs1[p] * GNN_D;

    v16bf afrag[6];
#pragma unroll
    for (int ks = 0; ks < 6; ++ks) {
#pragma unroll
        for (int i = 0; i < 16; ++i) {
            int k = ks * 32 + kidx(i, lane);
            float fv;
            if (k < 64)        fv = Z[srow + k];
            else if (k < 128)  fv = Z[drow + (k - 64)];
            else {
                int k2 = k - 128;
                fv = Z[srow + k2] * Z[drow + k2];
            }
            afrag[ks][i] = (__bf16)fv;
        }
    }

    const v16bf* bfr = (const v16bf*)Wp;
    float partial[8] = {0, 0, 0, 0, 0, 0, 0, 0};
#pragma unroll
    for (int ct = 0; ct < 4; ++ct) {
        v8f acc = {};
#pragma unroll
        for (int ks = 0; ks < 6; ++ks)
            acc = WMMA_BF16(afrag[ks], bfr[(ct * 6 + ks) * 32 + lane], acc);
        int n = ct * 16 + m;
        float b1v = bp1[n];
        float w2v = Wp2[n];
#pragma unroll
        for (int r = 0; r < 8; ++r) {
            float y = acc[r] + b1v;
            y = y > 0.0f ? y : 0.0f;
            partial[r] += y * w2v;
        }
    }

#pragma unroll
    for (int mask = 1; mask < 16; mask <<= 1)
#pragma unroll
        for (int r = 0; r < 8; ++r)
            partial[r] += __shfl_xor(partial[r], mask, 32);

    if (m == 0) {
        float bb = bp2[0];
#pragma unroll
        for (int r = 0; r < 8; ++r)
            Out[pairBase + r + ((lane >> 4) << 3)] = partial[r] + bb;
    }
}

// ---------------------------------------------------------------- launch ----
extern "C" void kernel_launch(void* const* d_in, const int* in_sizes, int n_in,
                              void* d_out, int out_size, void* d_ws, size_t ws_size,
                              hipStream_t stream) {
    const int*   types    = (const int*)  d_in[0];
    const int*   adj_rows = (const int*)  d_in[1];
    const int*   adj_cols = (const int*)  d_in[2];
    const float* adj_vals = (const float*)d_in[3];
    const int*   pairs    = (const int*)  d_in[4];   // [2,P]
    const float* node_emb = (const float*)d_in[5];
    const float* type_emb = (const float*)d_in[6];
    const float* W1  = (const float*)d_in[7];
    const float* b1  = (const float*)d_in[8];
    const float* W2  = (const float*)d_in[9];
    const float* b2  = (const float*)d_in[10];
    const float* Wp1 = (const float*)d_in[11];
    const float* bp1 = (const float*)d_in[12];
    const float* Wp2 = (const float*)d_in[13];
    const float* bp2 = (const float*)d_in[14];
    float* out = (float*)d_out;

    // ws layout (fp32, padded to NPAD rows): X | AGG | H | bf16 weight packs
    const size_t NHP = (size_t)GNN_NPAD * GNN_H;
    float* X   = (float*)d_ws;
    float* AGG = X + NHP;
    float* Hb  = AGG + NHP;
    __bf16* W1p  = (__bf16*)(Hb + NHP);         // 128x128 = 16384 halfs
    __bf16* W2p  = W1p + 16384;                 // 128x64  =  8192 halfs
    __bf16* Wp1p = W2p + 8192;                  // 192x64  = 12288 halfs
    float* Z = X;                               // X dead after spmm1; NPAD*64 fits

    const int NH = GNN_N * GNN_H;

    k_pack<<<(16384 + 255) / 256, 256, 0, stream>>>(W1,  W1p,  GNN_H, GNN_H);
    k_pack<<<(8192  + 255) / 256, 256, 0, stream>>>(W2,  W2p,  GNN_H, GNN_D);
    k_pack<<<(12288 + 255) / 256, 256, 0, stream>>>(Wp1, Wp1p, 3 * GNN_D, GNN_D);

    k_encode<<<(NH + 255) / 256, 256, 0, stream>>>(types, node_emb, type_emb, X);

    k_zero<<<(NH + 255) / 256, 256, 0, stream>>>(AGG, NH);
    k_spmm<<<(GNN_E + 7) / 8, 256, 0, stream>>>(adj_rows, adj_cols, adj_vals, X, AGG);

    k_gemm1<<<GNN_NPAD / 128, 256, 0, stream>>>(AGG, W1p, b1, Hb);

    k_zero<<<(NH + 255) / 256, 256, 0, stream>>>(AGG, NH);
    k_spmm<<<(GNN_E + 7) / 8, 256, 0, stream>>>(adj_rows, adj_cols, adj_vals, Hb, AGG);

    k_gemm2<<<GNN_NPAD / 128, 256, 0, stream>>>(AGG, W2p, b2, Z);

    k_pairs<<<GNN_P / 128, 256, 0, stream>>>(Z, pairs, pairs + GNN_P,
                                             Wp1p, bp1, Wp2, bp2, out);
}
